// HiDAN_58265526338187
// MI455X (gfx1250) — matrix operand order repaired and checked by reading, not compiled, not run on hardware
//
#include <hip/hip_runtime.h>
#include <hip/hip_bf16.h>
#include <stdint.h>

// ---------------------------------------------------------------------------
// HiDAN forward for MI455X (gfx1250, wave32, WMMA).
// bf16 WMMA (v_wmma_f32_16x16x32_bf16) with fp32 accumulate for all GEMMs.
// 128x128 block tiles, K-step 64, packed bf16 LDS staging, global prefetch.
// ---------------------------------------------------------------------------

typedef __attribute__((ext_vector_type(16))) __bf16 v16bf;
typedef __attribute__((ext_vector_type(8)))  float  v8f;

union FragAB {
  unsigned int u[8];
  v16bf v;
};

__device__ __forceinline__ unsigned short f2bf(float f) {
  unsigned int u = __float_as_uint(f);
  u += 0x7FFFu + ((u >> 16) & 1u);   // round-to-nearest-even
  return (unsigned short)(u >> 16);
}
__device__ __forceinline__ unsigned int pack2bf(float lo, float hi) {
  return (unsigned int)f2bf(lo) | ((unsigned int)f2bf(hi) << 16);
}
__device__ __forceinline__ float eluf(float x) { return x > 0.f ? x : (__expf(x) - 1.f); }
__device__ __forceinline__ v8f zero8() { v8f z = {0.f,0.f,0.f,0.f,0.f,0.f,0.f,0.f}; return z; }

#define BS_ 32
#define SL_ 512
#define H_  1024

// A-fragment K offset pattern within a 32-deep block (ISA 7.12.2, 16-bit A)
__device__ __forceinline__ int akoff(int g, int half) {
  return 2 * (g & 3) + ((g >> 2) << 4) + (half << 3);
}
// B-fragment K offset pattern within a 32-deep block (16-bit B, [n][k] LDS)
__device__ __forceinline__ int bkoff(int g, int half) {
  return 2 * (g & 3) + ((g >> 2) << 4) + (half << 3);
}

// ===========================================================================
// Generic GEMM: C[M,N] = act(A[M,K] @ W[K,N] + bias[N]) [* rowmask[row]]
// Block tile 128x128, 8 waves (4x2), wave tile 32x64 (2x4 frags), K step 64.
// ===========================================================================
__global__ void __launch_bounds__(256)
wmma_gemm_kernel(const float* __restrict__ A, const float* __restrict__ W,
                 const float* __restrict__ bias, const float* __restrict__ rowmask,
                 float* __restrict__ C, int M, int N, int K, int act)
{
  __shared__ unsigned short As[128][66];   // [m][k], padded stride
  __shared__ unsigned short Bs[128][66];   // [n][k] (W transposed)

  const int tid  = threadIdx.x;
  const int wave = tid >> 5, lane = tid & 31;
  const int wm   = wave >> 1, wn = wave & 1;
  const int half = lane >> 4, l16 = lane & 15;
  const int bm   = blockIdx.y * 128, bn = blockIdx.x * 128;

  v8f c[2][4];
  for (int i = 0; i < 2; ++i) for (int j = 0; j < 4; ++j) c[i][j] = zero8();

  for (int k0 = 0; k0 < K; k0 += 64) {
    __syncthreads();
    // stage A tile 128x64: float2 loads -> packed bf16 dword stores
    for (int i = 0; i < 16; ++i) {
      int idx = i * 256 + tid;
      int r = idx >> 5, cp = idx & 31;           // 32 pairs per row
      const float2 f = *(const float2*)&A[(size_t)(bm + r) * K + (k0 + 2 * cp)];
      *(unsigned int*)&As[r][2 * cp] = pack2bf(f.x, f.y);
    }
    // stage W tile 64x128 transposed: pack K-pairs per column
    for (int i = 0; i < 16; ++i) {
      int idx = i * 256 + tid;
      int n = idx & 127, kp = idx >> 7;          // kp 0..31
      size_t g = (size_t)(k0 + 2 * kp) * N + (bn + n);
      *(unsigned int*)&Bs[n][2 * kp] = pack2bf(W[g], W[g + N]);
    }
    if (k0 + 64 < K) {                           // speculative prefetch of next tiles
      __builtin_prefetch(&A[(size_t)(bm + (tid >> 1)) * K + (k0 + 64 + (tid & 1) * 32)], 0, 0);
      __builtin_prefetch(&W[(size_t)(k0 + 64 + (tid >> 2)) * N + (bn + (tid & 3) * 32)], 0, 0);
    }
    __syncthreads();

    for (int kb = 0; kb < 64; kb += 32) {
      FragAB a[2], b[4];
      for (int mf = 0; mf < 2; ++mf)
        for (int g = 0; g < 8; ++g)
          a[mf].u[g] = *(const unsigned int*)&As[wm * 32 + mf * 16 + l16][kb + akoff(g, half)];
      for (int nf = 0; nf < 4; ++nf)
        for (int g = 0; g < 8; ++g)
          b[nf].u[g] = *(const unsigned int*)&Bs[wn * 64 + nf * 16 + l16][kb + bkoff(g, half)];
      for (int mf = 0; mf < 2; ++mf)
        for (int nf = 0; nf < 4; ++nf)
          c[mf][nf] = __builtin_amdgcn_wmma_f32_16x16x32_bf16(
              false, a[mf].v, false, b[nf].v, (short)0, c[mf][nf], false, false);
    }
  }

  // epilogue: bias + optional ELU + optional row mask
  for (int mf = 0; mf < 2; ++mf)
    for (int nf = 0; nf < 4; ++nf) {
      int col = bn + wn * 64 + nf * 16 + l16;
      float bv = bias[col];
      for (int v = 0; v < 8; ++v) {
        int row = bm + wm * 32 + mf * 16 + half * 8 + v;
        float x = c[mf][nf][v] + bv;
        if (act) x = eluf(x);
        if (rowmask) x *= rowmask[row];
        C[(size_t)row * N + col] = x;
      }
    }
}

// ===========================================================================
// Gate GEMM: enc = (g*h + (1-g)*d)*mask, g = sigmoid([h|d] @ Wg + bg)
// Virtual concat A: k<1024 -> hidden, else depend. Same 128x128 tiling.
// ===========================================================================
__global__ void __launch_bounds__(256)
gate_gemm_kernel(const float* __restrict__ Hid, const float* __restrict__ Dep,
                 const float* __restrict__ Wg, const float* __restrict__ bg,
                 const float* __restrict__ rowmask, float* __restrict__ Enc)
{
  __shared__ unsigned short As[128][66];
  __shared__ unsigned short Bs[128][66];

  const int tid  = threadIdx.x;
  const int wave = tid >> 5, lane = tid & 31;
  const int wm   = wave >> 1, wn = wave & 1;
  const int half = lane >> 4, l16 = lane & 15;
  const int bm   = blockIdx.y * 128, bn = blockIdx.x * 128;
  const int N = H_, K = 2 * H_;

  v8f c[2][4];
  for (int i = 0; i < 2; ++i) for (int j = 0; j < 4; ++j) c[i][j] = zero8();

  for (int k0 = 0; k0 < K; k0 += 64) {
    __syncthreads();
    const float* src = (k0 < H_) ? Hid : Dep;      // 64 | 1024, tile never splits
    int kbase = (k0 < H_) ? k0 : (k0 - H_);
    for (int i = 0; i < 16; ++i) {
      int idx = i * 256 + tid;
      int r = idx >> 5, cp = idx & 31;
      const float2 f = *(const float2*)&src[(size_t)(bm + r) * H_ + (kbase + 2 * cp)];
      *(unsigned int*)&As[r][2 * cp] = pack2bf(f.x, f.y);
    }
    for (int i = 0; i < 16; ++i) {
      int idx = i * 256 + tid;
      int n = idx & 127, kp = idx >> 7;
      size_t g = (size_t)(k0 + 2 * kp) * N + (bn + n);
      *(unsigned int*)&Bs[n][2 * kp] = pack2bf(Wg[g], Wg[g + N]);
    }
    if (k0 + 64 < K)
      __builtin_prefetch(&Wg[(size_t)(k0 + 64 + (tid >> 2)) * N + (bn + (tid & 3) * 32)], 0, 0);
    __syncthreads();

    for (int kb = 0; kb < 64; kb += 32) {
      FragAB a[2], b[4];
      for (int mf = 0; mf < 2; ++mf)
        for (int g = 0; g < 8; ++g)
          a[mf].u[g] = *(const unsigned int*)&As[wm * 32 + mf * 16 + l16][kb + akoff(g, half)];
      for (int nf = 0; nf < 4; ++nf)
        for (int g = 0; g < 8; ++g)
          b[nf].u[g] = *(const unsigned int*)&Bs[wn * 64 + nf * 16 + l16][kb + bkoff(g, half)];
      for (int mf = 0; mf < 2; ++mf)
        for (int nf = 0; nf < 4; ++nf)
          c[mf][nf] = __builtin_amdgcn_wmma_f32_16x16x32_bf16(
              false, a[mf].v, false, b[nf].v, (short)0, c[mf][nf], false, false);
    }
  }

  for (int mf = 0; mf < 2; ++mf)
    for (int nf = 0; nf < 4; ++nf) {
      int col = bn + wn * 64 + nf * 16 + l16;
      float bv = bg[col];
      for (int v = 0; v < 8; ++v) {
        int row = bm + wm * 32 + mf * 16 + half * 8 + v;
        float g = 1.f / (1.f + __expf(-(c[mf][nf][v] + bv)));
        size_t off = (size_t)row * H_ + col;
        float e = (g * Hid[off] + (1.f - g) * Dep[off]) * rowmask[row];
        Enc[off] = e;
      }
    }
}

// ===========================================================================
// Fused attention: per block = (batch b, 16 query rows).
// Phase 1: logits[16][512] = head_i @ tail^T  (WMMA, logits kept in LDS)
// Phase 2: causal+length masked softmax in LDS, score -> bf16 in LDS
// Phase 3: depend[16][1024] = score @ hidden  (WMMA, score read from LDS)
// Logits/score never touch HBM. Static LDS ~59 KB (WGP has 320 KB).
// ===========================================================================
__global__ void __launch_bounds__(256)
attn_kernel(const float* __restrict__ head, const float* __restrict__ tail,
            const float* __restrict__ hidden, const float* __restrict__ mask,
            float* __restrict__ depend)
{
  __shared__ float          Lg[16][512];     // 32 KB logits
  __shared__ unsigned short Sc[16][520];     // 16.6 KB bf16 score (padded stride)
  __shared__ unsigned short As2[16][34];
  __shared__ unsigned short Bs2[128][34];
  __shared__ float          red[16][16];

  const int b   = blockIdx.y;
  const int i0  = blockIdx.x * 16;
  const int tid = threadIdx.x;
  const int wave = tid >> 5, lane = tid & 31;
  const int half = lane >> 4, l16 = lane & 15;
  const size_t rowb = (size_t)b * SL_;

  // ---------------- Phase 1: logits ----------------
  for (int jc = 0; jc < 4; ++jc) {           // 4 chunks of 128 cols; wave -> 16 cols
    v8f cl = zero8();
    for (int k0 = 0; k0 < H_; k0 += 32) {
      __syncthreads();
      {                                      // head tile 16x32, packed pairs
        int r = tid >> 4, cp = tid & 15;
        const float2 f = *(const float2*)&head[(rowb + i0 + r) * H_ + (k0 + 2 * cp)];
        *(unsigned int*)&As2[r][2 * cp] = pack2bf(f.x, f.y);
      }
      for (int i = 0; i < 8; ++i) {          // tail tile 128x32 -> [j][k], packed
        int idx = i * 256 + tid;
        int jr = idx >> 4, cp = idx & 15;
        const float2 f = *(const float2*)&tail[(rowb + jc * 128 + jr) * H_ + (k0 + 2 * cp)];
        *(unsigned int*)&Bs2[jr][2 * cp] = pack2bf(f.x, f.y);
      }
      if (k0 + 32 < H_)
        __builtin_prefetch(&tail[(rowb + jc * 128 + (tid >> 1)) * H_ + (k0 + 32 + (tid & 1) * 16)], 0, 0);
      __syncthreads();
      FragAB a, bb;
      for (int g = 0; g < 8; ++g)
        a.u[g] = *(const unsigned int*)&As2[l16][akoff(g, half)];
      for (int g = 0; g < 8; ++g)
        bb.u[g] = *(const unsigned int*)&Bs2[wave * 16 + l16][bkoff(g, half)];
      cl = __builtin_amdgcn_wmma_f32_16x16x32_bf16(false, a.v, false, bb.v,
                                                   (short)0, cl, false, false);
    }
    for (int v = 0; v < 8; ++v)
      Lg[half * 8 + v][jc * 128 + wave * 16 + l16] = cl[v];
  }
  __syncthreads();

  // ---------------- Phase 2: masked softmax ----------------
  {
    const int row = tid & 15, tcol = tid >> 4;   // 16 threads per logit row
    const int ig = i0 + row;
    float mmax = -3.0e38f;
    for (int k = 0; k < 32; ++k) {
      int j = tcol + (k << 4);
      bool valid = (ig > j) && (mask[rowb + j] > 0.f);
      float l = valid ? Lg[row][j] : -1.0e30f;
      Lg[row][j] = l;
      mmax = fmaxf(mmax, l);
    }
    red[row][tcol] = mmax; __syncthreads();
    for (int s = 8; s > 0; s >>= 1) {
      if (tcol < s) red[row][tcol] = fmaxf(red[row][tcol], red[row][tcol + s]);
      __syncthreads();
    }
    float rmax = red[row][0]; __syncthreads();
    float lsum = 0.f;
    for (int k = 0; k < 32; ++k) {
      int j = tcol + (k << 4);
      float p = __expf(Lg[row][j] - rmax);
      Lg[row][j] = p;
      lsum += p;
    }
    red[row][tcol] = lsum; __syncthreads();
    for (int s = 8; s > 0; s >>= 1) {
      if (tcol < s) red[row][tcol] += red[row][tcol + s];
      __syncthreads();
    }
    float inv = 1.f / red[row][0]; __syncthreads();
    for (int k = 0; k < 32; ++k) {
      int j = tcol + (k << 4);
      bool valid = (ig > j) && (mask[rowb + j] > 0.f);
      Sc[row][j] = f2bf(valid ? Lg[row][j] * inv : 0.f);
    }
  }
  __syncthreads();

  // ---------------- Phase 3: depend = score @ hidden ----------------
  for (int nc = 0; nc < 8; ++nc) {             // 8 chunks of 128 output cols
    v8f cd = zero8();
    for (int kt = 0; kt < 16; ++kt) {          // K = 512 (j dim), step 32
      int k0 = kt * 32;
      __syncthreads();
      for (int i = 0; i < 8; ++i) {            // hidden tile 32x128 -> [n][k], packed
        int idx = i * 256 + tid;
        int nn = idx & 127, kp = idx >> 7;     // kp 0..15
        size_t g = (rowb + k0 + 2 * kp) * H_ + (nc * 128 + nn);
        *(unsigned int*)&Bs2[nn][2 * kp] = pack2bf(hidden[g], hidden[g + H_]);
      }
      if (kt + 1 < 16)
        __builtin_prefetch(&hidden[(rowb + k0 + 32 + (tid >> 2)) * H_ + (nc * 128 + (tid & 3) * 32)], 0, 0);
      __syncthreads();
      FragAB a, bb;
      for (int g = 0; g < 8; ++g)
        a.u[g] = *(const unsigned int*)&Sc[l16][k0 + akoff(g, half)];
      for (int g = 0; g < 8; ++g)
        bb.u[g] = *(const unsigned int*)&Bs2[wave * 16 + l16][bkoff(g, half)];
      cd = __builtin_amdgcn_wmma_f32_16x16x32_bf16(false, a.v, false, bb.v,
                                                   (short)0, cd, false, false);
    }
    for (int v = 0; v < 8; ++v)
      depend[(rowb + i0 + half * 8 + v) * H_ + (nc * 128 + wave * 16 + l16)] = cd[v];
  }
}

// ===========================================================================
// Final pooling: map2 = (map1*time_inf)@Wm2+bm2 ; a = softmax_sl(masked)*mask;
// out = a * encoding. One block per batch.
// ===========================================================================
__global__ void __launch_bounds__(256)
final_kernel(const float* __restrict__ map1, const float* __restrict__ ti,
             const float* __restrict__ enc, const float* __restrict__ mask,
             const float* __restrict__ Wm2, const float* __restrict__ bm2,
             float* __restrict__ out)
{
  __shared__ float m2s[SL_];
  __shared__ float redf[8];
  const int b = blockIdx.x;
  const int tid = threadIdx.x, wave = tid >> 5, lane = tid & 31;
  const size_t rowb = (size_t)b * SL_;

  // per-row dot over H of map1*ti*Wm2
  for (int r = 0; r < 64; ++r) {
    int s = wave + (r << 3);
    const float* p1 = map1 + (rowb + s) * H_;
    const float* p2 = ti   + (rowb + s) * H_;
    float acc = 0.f;
    for (int h = lane; h < H_; h += 32) acc += p1[h] * p2[h] * Wm2[h];
    for (int off = 16; off > 0; off >>= 1) acc += __shfl_down(acc, off, 32);
    if (lane == 0) m2s[s] = acc + bm2[0];
  }
  __syncthreads();

  // masked softmax over s
  float lm = -3.0e38f;
  for (int s = tid; s < SL_; s += 256) {
    float v = m2s[s] + (-1.0e30f) * (1.f - mask[rowb + s]);
    m2s[s] = v;
    lm = fmaxf(lm, v);
  }
  for (int off = 16; off > 0; off >>= 1) lm = fmaxf(lm, __shfl_down(lm, off, 32));
  if (lane == 0) redf[wave] = lm;
  __syncthreads();
  if (tid == 0) {
    float m = redf[0];
    for (int w = 1; w < 8; ++w) m = fmaxf(m, redf[w]);
    redf[0] = m;
  }
  __syncthreads();
  float gmax = redf[0];
  __syncthreads();                             // everyone has read gmax
  float ls = 0.f;
  for (int s = tid; s < SL_; s += 256) {
    float p = __expf(m2s[s] - gmax);
    m2s[s] = p;
    ls += p;
  }
  for (int off = 16; off > 0; off >>= 1) ls += __shfl_down(ls, off, 32);
  if (lane == 0) redf[wave] = ls;
  __syncthreads();
  if (tid == 0) {
    float ssum = 0.f;
    for (int w = 0; w < 8; ++w) ssum += redf[w];
    redf[0] = ssum;
  }
  __syncthreads();
  float inv = 1.f / redf[0];
  for (int s = tid; s < SL_; s += 256)
    m2s[s] = m2s[s] * inv * mask[rowb + s];
  __syncthreads();

  // out = a * encoding
  for (int i = 0; i < (SL_ * H_) / 256; ++i) {
    int idx = i * 256 + tid;
    int s = idx >> 10, h = idx & (H_ - 1);
    size_t off = (rowb + s) * H_ + h;
    out[off] = m2s[s] * enc[off];
  }
}

// ===========================================================================
extern "C" void kernel_launch(void* const* d_in, const int* in_sizes, int n_in,
                              void* d_out, int out_size, void* d_ws, size_t ws_size,
                              hipStream_t stream)
{
  (void)in_sizes; (void)n_in; (void)out_size; (void)ws_size;
  const float* cas_emb  = (const float*)d_in[0];   // [32,512,1024]
  const float* cas_mask = (const float*)d_in[1];   // [32,512,1] -> row mask
  const float* time_w   = (const float*)d_in[2];   // [32,512,64]
  const float* W1  = (const float*)d_in[3];  const float* b1  = (const float*)d_in[4];
  const float* Wh  = (const float*)d_in[5];  const float* bh  = (const float*)d_in[6];
  const float* Wt  = (const float*)d_in[7];  const float* bt  = (const float*)d_in[8];
  const float* Wg  = (const float*)d_in[9];  const float* bg  = (const float*)d_in[10];
  const float* Wm1 = (const float*)d_in[11]; const float* bm1 = (const float*)d_in[12];
  const float* Wti = (const float*)d_in[13]; const float* bti = (const float*)d_in[14];
  const float* Wm2 = (const float*)d_in[15]; const float* bm2 = (const float*)d_in[16];

  const int M = BS_ * SL_;                   // 16384
  const size_t SZ = (size_t)M * H_;          // 16.78M floats = 64 MB
  float* ws     = (float*)d_ws;
  float* hidden = ws;                        // cas_hidden        (later: time_inf)
  float* head   = ws + SZ;                   // head              (later: encoding)
  float* tail   = ws + 2 * SZ;               // tail              (later: map1)
  float* dep    = ws + 3 * SZ;               // depend

  dim3 blk(256);
  dim3 gemm_grid(H_ / 128, M / 128);         // (8, 128)

  // 1) cas_hidden = elu(emb@W1+b1) * mask
  wmma_gemm_kernel<<<gemm_grid, blk, 0, stream>>>(cas_emb, W1, b1, cas_mask,
                                                  hidden, M, H_, H_, 1);
  // 2) head / tail
  wmma_gemm_kernel<<<gemm_grid, blk, 0, stream>>>(hidden, Wh, bh, nullptr,
                                                  head, M, H_, H_, 0);
  wmma_gemm_kernel<<<gemm_grid, blk, 0, stream>>>(hidden, Wt, bt, nullptr,
                                                  tail, M, H_, H_, 0);
  // 3) fused masked attention -> depend
  attn_kernel<<<dim3(SL_ / 16, BS_), blk, 0, stream>>>(head, tail, hidden,
                                                       cas_mask, dep);
  // 4) gate + combine -> encoding (reuses head buffer)
  float* enc = head;
  gate_gemm_kernel<<<gemm_grid, blk, 0, stream>>>(hidden, dep, Wg, bg,
                                                  cas_mask, enc);
  // 5) map1 = elu(enc@Wm1+bm1) (reuses tail); time_inf = elu(tw@Wti+bti) (reuses hidden)
  float* map1 = tail;
  wmma_gemm_kernel<<<gemm_grid, blk, 0, stream>>>(enc, Wm1, bm1, nullptr,
                                                  map1, M, H_, H_, 1);
  float* tinf = hidden;
  wmma_gemm_kernel<<<gemm_grid, blk, 0, stream>>>(time_w, Wti, bti, nullptr,
                                                  tinf, M, H_, 64, 1);
  // 6) pooling softmax + scale -> out
  final_kernel<<<dim3(BS_), blk, 0, stream>>>(map1, tinf, enc, cas_mask,
                                              Wm2, bm2, (float*)d_out);
}
